// ExperimentLinear_15324443312229
// MI455X (gfx1250) — 8-bit matrix operand order repaired and checked by reading, compiled, not run on hardware
//
#include <hip/hip_runtime.h>
#include <stdint.h>

typedef __attribute__((ext_vector_type(8))) int v8i;

#define QMAXF 127.0f
#define EPSF  1e-8f

__global__ __launch_bounds__(256) void quant_rows_kernel(
    const float* __restrict__ src, int8_t* __restrict__ q,
    float* __restrict__ scales, int K)
{
    const int row = blockIdx.x;
    const int tid = threadIdx.x;
    const float4* r4 = reinterpret_cast<const float4*>(src + (size_t)row * K);
    const int vpt = K >> 10;

    float amax = 0.0f;
#pragma unroll
    for (int i = 0; i < 4; ++i) {
        if (i < vpt) {
            float4 v = r4[tid + i * 256];
            amax = fmaxf(amax, fmaxf(fmaxf(fabsf(v.x), fabsf(v.y)),
                                     fmaxf(fabsf(v.z), fabsf(v.w))));
        }
    }

    __shared__ float red[256];
    red[tid] = amax;
    __syncthreads();
#pragma unroll
    for (int s = 128; s > 0; s >>= 1) {
        if (tid < s) red[tid] = fmaxf(red[tid], red[tid + s]);
        __syncthreads();
    }
    const float scale = fmaxf(red[0] * (1.0f / QMAXF), EPSF);
    if (tid == 0) scales[row] = scale;

    uint32_t* qrow = reinterpret_cast<uint32_t*>(q + (size_t)row * K);
#pragma unroll
    for (int i = 0; i < 4; ++i) {
        if (i < vpt) {
            float4 v = r4[tid + i * 256];
            int b0 = (int)fminf(fmaxf(rintf(v.x / scale), -128.0f), 127.0f);
            int b1 = (int)fminf(fmaxf(rintf(v.y / scale), -128.0f), 127.0f);
            int b2 = (int)fminf(fmaxf(rintf(v.z / scale), -128.0f), 127.0f);
            int b3 = (int)fminf(fmaxf(rintf(v.w / scale), -128.0f), 127.0f);
            uint32_t packed = (uint32_t)(uint8_t)b0 |
                              ((uint32_t)(uint8_t)b1 << 8) |
                              ((uint32_t)(uint8_t)b2 << 16) |
                              ((uint32_t)(uint8_t)b3 << 24);
            qrow[tid + i * 256] = packed;
        }
    }
}

__device__ __forceinline__ void async_copy_b128(uint32_t ldsOff, const int8_t* g)
{
    asm volatile("global_load_async_to_lds_b128 %0, %1, off"
                 :: "v"(ldsOff), "v"(g) : "memory");
}

#define KSTEP      128
#define LDS_STRIDE 136

__device__ __forceinline__ void stage_compute(
    const int8_t* aAddr,
    const int8_t* b0, const int8_t* b1, const int8_t* b2, const int8_t* b3,
    v8i acc[4])
{
    const int8_t* bp[4] = {b0, b1, b2, b3};

#pragma unroll
    for (int h = 0; h < 2; ++h) {
        v8i a;
#pragma unroll
        for (int j = 0; j < 4; ++j) {
            uint2 d = *reinterpret_cast<const uint2*>(aAddr + h * 64 + j * 16);
            a[2 * j]     = (int)d.x;
            a[2 * j + 1] = (int)d.y;
        }
        v8i bf[4];
#pragma unroll
        for (int ct = 0; ct < 4; ++ct) {
#pragma unroll
            for (int j = 0; j < 4; ++j) {
                uint2 d = *reinterpret_cast<const uint2*>(
                    bp[ct] + h * 64 + j * 16);
                bf[ct][2 * j]     = (int)d.x;
                bf[ct][2 * j + 1] = (int)d.y;
            }
        }
#pragma unroll
        for (int ct = 0; ct < 4; ++ct) {
            acc[ct] = __builtin_amdgcn_wmma_i32_16x16x64_iu8(
                true, a, true, bf[ct], acc[ct], false, false);
        }
    }
}

__global__ __launch_bounds__(256) void gemm_i8_wmma_kernel(
    const int8_t* __restrict__ qx, const float* __restrict__ sx,
    const int8_t* __restrict__ qw, const float* __restrict__ sw,
    const float* __restrict__ bias, float* __restrict__ out,
    int M, int N, int K)
{
    __shared__ int8_t As[2][128 * LDS_STRIDE];
    __shared__ int8_t Bs[2][64 * LDS_STRIDE];

    const int tid  = threadIdx.x;
    const int lane = tid & 31;
    const int wave = tid >> 5;
    const int l15  = lane & 15;
    const int hi   = lane >> 4;

    const int mBlk  = blockIdx.y * 128;
    const int nBase = blockIdx.x * 64;

    const int crow = tid >> 3;
    const int cseg = (tid & 7) * 16;

    const int8_t* gA[4];
    const int8_t* gB[2];
#pragma unroll
    for (int i = 0; i < 4; ++i)
        gA[i] = qx + (size_t)(mBlk + crow + 32 * i) * K + cseg;
#pragma unroll
    for (int i = 0; i < 2; ++i)
        gB[i] = qw + (size_t)(nBase + crow + 32 * i) * K + cseg;

    uint32_t lA[2][4], lB[2][2];
#pragma unroll
    for (int s = 0; s < 2; ++s) {
#pragma unroll
        for (int i = 0; i < 4; ++i)
            lA[s][i] = (uint32_t)(uintptr_t)
                &As[s][(crow + 32 * i) * LDS_STRIDE + cseg];
#pragma unroll
        for (int i = 0; i < 2; ++i)
            lB[s][i] = (uint32_t)(uintptr_t)
                &Bs[s][(crow + 32 * i) * LDS_STRIDE + cseg];
    }

    const int8_t* aAddr[2];
    const int8_t* bAddr[2][4];
#pragma unroll
    for (int s = 0; s < 2; ++s) {
        aAddr[s] = &As[s][(wave * 16 + l15) * LDS_STRIDE + hi * 8];
#pragma unroll
        for (int ct = 0; ct < 4; ++ct)
            bAddr[s][ct] = &Bs[s][(ct * 16 + l15) * LDS_STRIDE + hi * 8];
    }

    v8i zero = {0, 0, 0, 0, 0, 0, 0, 0};
    v8i acc[4];
    acc[0] = zero; acc[1] = zero; acc[2] = zero; acc[3] = zero;

#pragma unroll
    for (int i = 0; i < 4; ++i) async_copy_b128(lA[0][i], gA[i]);
#pragma unroll
    for (int i = 0; i < 2; ++i) async_copy_b128(lB[0][i], gB[i]);

    const int nk = K / KSTEP;
    for (int ks = 0; ks < nk; ks += 2) {
        {
            const int kn = (ks + 1) * KSTEP;
#pragma unroll
            for (int i = 0; i < 4; ++i) async_copy_b128(lA[1][i], gA[i] + kn);
#pragma unroll
            for (int i = 0; i < 2; ++i) async_copy_b128(lB[1][i], gB[i] + kn);
            asm volatile("s_wait_asynccnt 0x6" ::: "memory");
        }
        __syncthreads();
        stage_compute(aAddr[0], bAddr[0][0], bAddr[0][1], bAddr[0][2],
                      bAddr[0][3], acc);
        __syncthreads();

        if (ks + 2 < nk) {
            const int kn = (ks + 2) * KSTEP;
#pragma unroll
            for (int i = 0; i < 4; ++i) async_copy_b128(lA[0][i], gA[i] + kn);
#pragma unroll
            for (int i = 0; i < 2; ++i) async_copy_b128(lB[0][i], gB[i] + kn);
            asm volatile("s_wait_asynccnt 0x6" ::: "memory");
        } else {
            asm volatile("s_wait_asynccnt 0x0" ::: "memory");
        }
        __syncthreads();
        stage_compute(aAddr[1], bAddr[1][0], bAddr[1][1], bAddr[1][2],
                      bAddr[1][3], acc);
        __syncthreads();
    }

    const int mBase = mBlk + wave * 16;
    float swv[4], bv[4];
#pragma unroll
    for (int ct = 0; ct < 4; ++ct) {
        const int n = nBase + ct * 16 + l15;
        swv[ct] = sw[n];
        bv[ct]  = bias[n];
    }
#pragma unroll
    for (int r = 0; r < 8; ++r) {
        const int m = mBase + hi * 8 + r;
        const float sxv = sx[m];
        float* orow = out + (size_t)m * N + nBase + l15;
#pragma unroll
        for (int ct = 0; ct < 4; ++ct) {
            orow[ct * 16] = (float)acc[ct][r] * (sxv * swv[ct]) + bv[ct];
        }
    }
}

extern "C" void kernel_launch(void* const* d_in, const int* in_sizes, int n_in,
                              void* d_out, int out_size, void* d_ws, size_t ws_size,
                              hipStream_t stream) {
    const float* x = (const float*)d_in[0];
    const float* w = (const float*)d_in[1];
    const float* b = (const float*)d_in[2];
    float* out = (float*)d_out;

    const int O = in_sizes[2];
    const int K = in_sizes[1] / O;
    const int M = in_sizes[0] / K;

    uint8_t* ws = (uint8_t*)d_ws;
    int8_t* qx = (int8_t*)ws;
    int8_t* qw = (int8_t*)(ws + (size_t)M * K);
    float*  sx = (float*)(ws + (size_t)M * K + (size_t)O * K);
    float*  sw = sx + M;

    quant_rows_kernel<<<M, 256, 0, stream>>>(x, qx, sx, K);
    quant_rows_kernel<<<O, 256, 0, stream>>>(w, qw, sw, K);

    dim3 grid(O / 64, M / 128);
    gemm_i8_wmma_kernel<<<grid, 256, 0, stream>>>(qx, sx, qw, sw, b, out, M, O, K);
}
